// MyModelONNX_43662637531519
// MI455X (gfx1250) — compile-verified, matrix operand order checked by reading, not run on hardware
//
#include <hip/hip_runtime.h>
#include <hip/hip_bf16.h>

typedef __attribute__((ext_vector_type(16))) _Float16 v16h;
typedef __attribute__((ext_vector_type(8)))  float    v8f;

#define NUM_HEADS 4
#define HD 32
#define NN 49
#define NPAD 64
#define NW 64
#define ATTN_SCALE 0.1767766952966369f
#define LOG2E 1.4426950408889634f

// LDS row strides (in halves) padded for conflict-free 16-lane fragment loads
#define QK_STRIDE 34   // 68B rows -> bank = 17*l mod 64, distinct over 16 lanes
#define VT_STRIDE 68   // 136B rows -> bank = 34*l mod 64, distinct over 16 lanes
#define AT_STRIDE 68

// ---------------------------------------------------------------------------
// WMMA 16-bit A-fragment loader (ISA 7.12.2), row-major source in LDS:
//   lanes 0-15:  M=l,    K in {0..7, 16..23}  (2 halves per VGPR)
//   lanes 16-31: M=l-16, K in {8..15, 24..31}
// B operands use the mirrored striping (B == A-layout of B^T), so row-major
// tiles of K (for Q*K^T) and of V^T (for P*V) both use this loader.
__device__ __forceinline__ v16h frag_load(const _Float16* base, int stride_h, int lane) {
  const int m  = lane & 15;
  const int kb = (lane & 16) ? 8 : 0;
  const _Float16* p = base + m * stride_h + kb;
  v16h r;
  uint32_t* rp = reinterpret_cast<uint32_t*>(&r);
#pragma unroll
  for (int j = 0; j < 4; ++j)
    rp[j] = *reinterpret_cast<const uint32_t*>(p + 2 * j);
#pragma unroll
  for (int j = 0; j < 4; ++j)
    rp[4 + j] = *reinterpret_cast<const uint32_t*>(p + 16 + 2 * j);
  return r;
}

// ---------------------------------------------------------------------------
// 16-lane butterfly reductions in pure DPP16 (no LDS, no index setup).
// quad_perm[1,0,3,2]=0xB1 (xor1), quad_perm[2,3,0,1]=0x4E (xor2),
// row_half_mirror=0x141 (pairs quads), row_mirror=0x140 (pairs 8-blocks).
// Lanes 0-15 and 16-31 reduce independently (DPP16 rows).
template <int CTRL>
__device__ __forceinline__ float dpp_perm(float v) {
  return __builtin_bit_cast(float, __builtin_amdgcn_update_dpp(
      0, __builtin_bit_cast(int, v), CTRL, 0xF, 0xF, true));
}
__device__ __forceinline__ float hmax16(float v) {
  v = fmaxf(v, dpp_perm<0xB1>(v));
  v = fmaxf(v, dpp_perm<0x4E>(v));
  v = fmaxf(v, dpp_perm<0x141>(v));
  v = fmaxf(v, dpp_perm<0x140>(v));
  return v;
}
__device__ __forceinline__ float hsum16(float v) {
  v += dpp_perm<0xB1>(v);
  v += dpp_perm<0x4E>(v);
  v += dpp_perm<0x141>(v);
  v += dpp_perm<0x140>(v);
  return v;
}

__device__ __forceinline__ uint32_t pack2h(float a, float b) {
  // v_cvt_pk_rtz_f16_f32 — one instruction for two converts + pack
  return __builtin_bit_cast(uint32_t, __builtin_amdgcn_cvt_pkrtz(a, b));
}

// ---------------------------------------------------------------------------
// Precompute comb[h][w][64][64] = log2e*(mask[h]+rot[w]) padded:
// cols >= 49 -> -inf (softmax kill), rows >= 49 -> 0 (discarded, kept finite).
// 256 combos reused by 16384 blocks; runs every launch (deterministic).
__global__ __launch_bounds__(256)
void build_comb_kernel(const float* __restrict__ mask,
                       const float* __restrict__ rot,
                       float* __restrict__ comb) {
  const int combo = blockIdx.x;          // h*64 + w
  const int h = combo >> 6;
  const int w = combo & (NW - 1);
  const float* mh = mask + (size_t)h * (NN * NN);
  const float* rw = rot  + (size_t)w * (NN * NN);
  float* dst = comb + (size_t)combo * (NPAD * NPAD);
  for (int i = threadIdx.x; i < NPAD * NPAD; i += 256) {
    const int M = i >> 6, N = i & 63;
    float v;
    if (N >= NN)      v = -INFINITY;
    else if (M >= NN) v = 0.0f;
    else              v = LOG2E * (mh[M * NN + N] + rw[M * NN + N]);
    dst[i] = v;
  }
}

// ---------------------------------------------------------------------------
template <bool USE_COMB>
__global__ __launch_bounds__(128)
void win_attn_wmma_kernel(const float* __restrict__ qkv,
                          const float* __restrict__ mask,
                          const float* __restrict__ rot,
                          float* __restrict__ out,
                          const float* __restrict__ comb) {
  __shared__ _Float16 qs[NUM_HEADS][NPAD][QK_STRIDE];  // Q * (scale*log2e), rows padded w/ 0
  __shared__ _Float16 ks[NUM_HEADS][NPAD][QK_STRIDE];  // K, rows padded w/ 0
  __shared__ _Float16 vT[NUM_HEADS][HD][VT_STRIDE];    // V transposed [d][k], k padded w/ 0
  __shared__ _Float16 at[NUM_HEADS][16][AT_STRIDE];    // per-WAVE (exclusive) P relayout scratch

  const int b    = blockIdx.x;
  const int tid  = threadIdx.x;
  const int wave = tid >> 5;
  const int lane = tid & 31;
  const int w    = b & (NW - 1);   // window index within image (B = imgs * 64)

  // ---- zero the K-dimension padding ----------------------------------------
  for (int i = tid; i < NUM_HEADS * 15 * 16; i += 128) {
    int h = i / (15 * 16);
    int r = 49 + (i / 16) % 15;
    int c = (i % 16) * 2;
    *reinterpret_cast<uint32_t*>(&qs[h][r][c]) = 0u;
    *reinterpret_cast<uint32_t*>(&ks[h][r][c]) = 0u;
  }
  for (int i = tid; i < NUM_HEADS * HD * 15; i += 128) {
    int h = i / (HD * 15);
    int d = (i / 15) % HD;
    int c = 49 + i % 15;
    vT[h][d][c] = (_Float16)0.0f;
  }

  // ---- stage qkv[b] (49x384 f32) -> LDS f16, one pass, float4 loads --------
  const float* src = qkv + (size_t)b * (NN * 384);
  const float qsc = ATTN_SCALE * LOG2E;   // fold exp2-domain factor into Q
  for (int i = tid; i < (NN * 384) / 4; i += 128) {
    int flat = i * 4;
    int n = flat / 384;
    int r = flat % 384;
    int s = r >> 7;          // 0=q 1=k 2=v
    int h = (r >> 5) & 3;
    int d = r & 31;          // multiple of 4
    float4 v4 = *reinterpret_cast<const float4*>(src + flat);
    if (s == 0) {
      uint32_t* qp = reinterpret_cast<uint32_t*>(&qs[h][n][d]);
      qp[0] = pack2h(v4.x * qsc, v4.y * qsc);
      qp[1] = pack2h(v4.z * qsc, v4.w * qsc);
    } else if (s == 1) {
      uint32_t* kp = reinterpret_cast<uint32_t*>(&ks[h][n][d]);
      kp[0] = pack2h(v4.x, v4.y);
      kp[1] = pack2h(v4.z, v4.w);
    } else {
      vT[h][d + 0][n] = (_Float16)v4.x;
      vT[h][d + 1][n] = (_Float16)v4.y;
      vT[h][d + 2][n] = (_Float16)v4.z;
      vT[h][d + 3][n] = (_Float16)v4.w;
    }
  }
  __syncthreads();   // the only cross-wave dependency

  // ---- per-wave attention for head h = wave --------------------------------
  const int h = wave;
  const int lhalf = lane >> 4;   // C/D fragment: vgpr r -> row r + 8*lhalf
  const int lcol  = lane & 15;   //               column = lane & 15

  // per-lane base into padded comb tile: all per-(mt,nt,r) offsets are
  // compile-time constants (stride 64) -> immediate-offset load clause
  const float* cb = USE_COMB
      ? comb + ((size_t)(h * NW + w) * (NPAD * NPAD) + (lhalf << 3) * NPAD + lcol)
      : nullptr;
  const float* mh = mask + (size_t)h * (NN * NN);
  const float* rw = rot  + (size_t)w * (NN * NN);

  v16h kf[4];
#pragma unroll
  for (int nt = 0; nt < 4; ++nt)
    kf[nt] = frag_load(&ks[h][nt * 16][0], QK_STRIDE, lane);

#pragma unroll
  for (int mt = 0; mt < 4; ++mt) {
    // S (exp2 domain) = (Q*scale*log2e) * K^T : K-dim = head_dim = 32 exactly
    v16h qa = frag_load(&qs[h][mt * 16][0], QK_STRIDE, lane);
    v8f acc[4];
#pragma unroll
    for (int nt = 0; nt < 4; ++nt) {
      v8f z = {};
      acc[nt] = __builtin_amdgcn_wmma_f32_16x16x32_f16(
          false, qa, false, kf[nt], (short)0, z, false, false);
    }

    // + log2e*(mask[h] + rotate_mask[w]) with padding pre-baked
    if (USE_COMB) {
#pragma unroll
      for (int nt = 0; nt < 4; ++nt) {
#pragma unroll
        for (int r = 0; r < 8; ++r) {
          acc[nt][r] += cb[(mt * 16 + r) * NPAD + nt * 16];
        }
      }
    } else {
      // fallback: branchless via index clamp; only cols N>=49 need -inf
#pragma unroll
      for (int nt = 0; nt < 4; ++nt) {
        const int Nc = (nt == 3) ? ((lcol > 0) ? NN - 1 : 48) : nt * 16 + lcol;
#pragma unroll
        for (int r = 0; r < 8; ++r) {
          const int M  = mt * 16 + r + (lhalf << 3);
          const int Mc = (mt == 3) ? ((M < NN) ? M : NN - 1) : M;
          const int o  = Mc * NN + Nc;
          float msum = mh[o] + rw[o];
          float v = fmaf(msum, LOG2E, acc[nt][r]);
          if (nt == 3) v = (lcol > 0) ? -INFINITY : v;
          acc[nt][r] = v;
        }
      }
    }

    // row softmax (exp2 domain): lanewise across tiles, then one DPP16 tree
    float inv[8];
#pragma unroll
    for (int r = 0; r < 8; ++r) {
      float m01  = fmaxf(fmaxf(acc[0][r], acc[1][r]),
                         fmaxf(acc[2][r], acc[3][r]));
      float rmax = hmax16(m01);
      float s = 0.0f;
#pragma unroll
      for (int nt = 0; nt < 4; ++nt) {
        float e = __builtin_amdgcn_exp2f(acc[nt][r] - rmax);
        acc[nt][r] = e;   // unnormalized; e <= 1, safe in f16
        s += e;
      }
      inv[r] = __builtin_amdgcn_rcpf(hsum16(s));  // applied to outputs later
    }

    // relayout P (C-fragment) -> row-major f16 in this wave's private scratch
#pragma unroll
    for (int nt = 0; nt < 4; ++nt) {
#pragma unroll
      for (int r = 0; r < 8; ++r) {
        at[h][r + (lhalf << 3)][nt * 16 + lcol] = (_Float16)acc[nt][r];
      }
    }
    // wave-private scratch: LDS ops of a wave are in-order in the LDS pipe,
    // so a dscnt drain (no block barrier) makes the writes readable.
    asm volatile("s_wait_dscnt 0" ::: "memory");

    // O_tile[16q x 32d] = P[16 x 64] * V[64 x 32] : 2 d-tiles x 2 K-chunks
#pragma unroll
    for (int dt = 0; dt < 2; ++dt) {
      v8f o = {};
#pragma unroll
      for (int kc = 0; kc < 2; ++kc) {
        v16h aa = frag_load(&at[h][0][kc * 32], AT_STRIDE, lane);
        v16h bb = frag_load(&vT[h][dt * 16][kc * 32], VT_STRIDE, lane);
        o = __builtin_amdgcn_wmma_f32_16x16x32_f16(
            false, aa, false, bb, (short)0, o, false, false);
      }
      // out[b][q][h*32 + d] with deferred 1/rowsum; rows q>=49 only in mt==3
#pragma unroll
      for (int r = 0; r < 8; ++r) {
        const int q = mt * 16 + r + (lhalf << 3);
        if (mt < 3 || q < NN) {
          out[((size_t)b * NN + q) * 128 + h * HD + dt * 16 + lcol] = o[r] * inv[r];
        }
      }
    }
    // keep this iteration's reads ordered before next iteration's overwrites
    asm volatile("s_wait_dscnt 0" ::: "memory");
  }
}

extern "C" void kernel_launch(void* const* d_in, const int* in_sizes, int n_in,
                              void* d_out, int out_size, void* d_ws, size_t ws_size,
                              hipStream_t stream) {
  const float* qkv  = (const float*)d_in[0];
  const float* mask = (const float*)d_in[1];
  const float* rot  = (const float*)d_in[2];
  // d_in[3] (hw_pad) is metadata only — unused in the math.
  float* out = (float*)d_out;
  const int B = in_sizes[0] / (NN * 384);   // 4096

  const size_t comb_bytes = (size_t)NUM_HEADS * NW * NPAD * NPAD * sizeof(float); // 4 MiB
  if (ws_size >= comb_bytes) {
    float* comb = (float*)d_ws;
    build_comb_kernel<<<NUM_HEADS * NW, 256, 0, stream>>>(mask, rot, comb);
    win_attn_wmma_kernel<true><<<B, 128, 0, stream>>>(qkv, mask, rot, out, comb);
  } else {
    win_attn_wmma_kernel<false><<<B, 128, 0, stream>>>(qkv, mask, rot, out, nullptr);
  }
}
